// GCN_Masked_46789373722952
// MI455X (gfx1250) — matrix operand order repaired and checked by reading, not compiled, not run on hardware
//
#include <hip/hip_runtime.h>
#include <hip/hip_bf16.h>
#include <cstdint>
#include <math.h>

#define FIN 128
#define HD  128
#define NC  40
#define BN_EPS 1e-5f

typedef __attribute__((ext_vector_type(16))) _Float16 v16h;
typedef __attribute__((ext_vector_type(8)))  float    v8f;

__device__ __forceinline__ float clamp01(float x) { return fminf(fmaxf(x, 0.f), 1.f); }

__device__ __forceinline__ uint32_t pkh(float a, float b) {
  union { _Float16 h[2]; uint32_t u; } c;
  c.h[0] = (_Float16)a; c.h[1] = (_Float16)b;
  return c.u;
}

// ---------------------------------------------------------------------------
// Pack masked weight W[K x ncols] (row-major) into WMMA B-fragment order:
//   idx = ((kk*CT + ct)*32 + lane)*16 + j
//   element: k = kk*32 + (lane>>4)*16 + j ; col = ct*16 + (lane&15)
// Columns >= ncols are zero-padded (layer 2: 40 cols padded to 48).
// ---------------------------------------------------------------------------
__global__ void pack_w(const float* __restrict__ W, const float* __restrict__ Wm,
                       _Float16* __restrict__ out, int ncols, int CT) {
  int tid = blockIdx.x * blockDim.x + threadIdx.x;
  int total = 4 * CT * 512;                 // K=128 -> 4 k-tiles
  if (tid >= total) return;
  int j    = tid & 15;
  int lane = (tid >> 4) & 31;
  int ct   = (tid >> 9) % CT;
  int kk   = tid / (512 * CT);
  int k    = kk * 32 + ((lane >> 4) << 4) + j;
  int col  = ct * 16 + (lane & 15);
  float v = 0.f;
  if (col < ncols) v = clamp01(Wm[k * ncols + col]) * W[k * ncols + col];
  out[tid] = (_Float16)v;
}

// ---------------------------------------------------------------------------
// WMMA GEMM:  Y[n x ncols_store] = X[n x 128] @ Wpacked  (f16 inputs, f32 acc)
// block = 256 threads (8 waves). Block stages a 16x128 row-tile of X into LDS
// as packed f16; wave w computes a 16x16 output tile for col tile w via
// v_wmma_f32_16x16x32_f16 (K=128 in 4 steps). The D tiles are bounced through
// LDS (aliased with the input staging buffer) so all global stores are
// coalesced b128.  n must be a multiple of 16 (N = 100000 = 6250*16).
// ---------------------------------------------------------------------------
__global__ __launch_bounds__(256) void gemm_wmma(
    const float* __restrict__ X, const _Float16* __restrict__ Bpk,
    float* __restrict__ Y, int ncols_store, int nctiles) {
  // aliased LDS: input = 16 x 68 u32 (4352 B), output = 16 x 132 f32 (8448 B)
  __shared__ __align__(16) char smem_raw[16 * 132 * 4];
  uint32_t (*sin)[68]  = (uint32_t(*)[68])smem_raw;
  float    (*sout)[132] = (float(*)[132])smem_raw;

  int row0 = blockIdx.x * 16;
  int t = threadIdx.x;
  int wave = t >> 5, lane = t & 31;

  // ---- stage 16x128 f32 -> packed f16 pairs (8 elements per thread) ----
  {
    int linear = t * 8;
    int r  = linear >> 7;
    int c0 = linear & 127;
    const float* src = X + (size_t)(row0 + r) * 128 + c0;
    float4 v0 = *(const float4*)(src);
    float4 v1 = *(const float4*)(src + 4);
    int p0 = c0 >> 1;
    sin[r][p0 + 0] = pkh(v0.x, v0.y);
    sin[r][p0 + 1] = pkh(v0.z, v0.w);
    sin[r][p0 + 2] = pkh(v1.x, v1.y);
    sin[r][p0 + 3] = pkh(v1.z, v1.w);
  }
  __syncthreads();

  // ---- WMMA: wave-uniform guard (EXEC all-ones inside) ----
  v8f acc = {};
  if (wave < nctiles) {
    int rrow = lane & 15;
    int kset = (lane >> 4) * 16;            // contiguous k order: k = 16h + s (matches B pack)
    const _Float16* bp = Bpk + (size_t)(wave * 32 + lane) * 16;
    size_t bstride = (size_t)nctiles * 32 * 16;

#pragma unroll
    for (int kk = 0; kk < 4; ++kk) {
      v16h a;
#pragma unroll
      for (int p = 0; p < 8; ++p) {
        int khalf = kk * 32 + kset + 2 * p;
        union { uint32_t u; _Float16 h[2]; } c;
        c.u = sin[rrow][khalf >> 1];
        a[2 * p]     = c.h[0];
        a[2 * p + 1] = c.h[1];
      }
      v16h b = *(const v16h*)(bp + (size_t)kk * bstride);
      acc = __builtin_amdgcn_wmma_f32_16x16x32_f16(
          false, a, false, b, (short)0, acc, false, false);
    }
  }
  __syncthreads();                          // all reads of sin done

  // ---- D tile -> LDS ----
  // D layout: VGPR i -> (M=i, N=lane) lanes 0-15 ; (M=i+8, N=lane-16) lanes 16-31
  if (wave < nctiles) {
    int col = wave * 16 + (lane & 15);
    int r0  = (lane >> 4) << 3;
#pragma unroll
    for (int i = 0; i < 8; ++i) sout[r0 + i][col] = acc[i];
  }
  __syncthreads();

  // ---- coalesced b128 stores ----
  if (ncols_store == 128) {
    int linear = t * 8;
    int r  = linear >> 7;
    int c0 = linear & 127;
    float4* dst = (float4*)(Y + (size_t)(row0 + r) * 128 + c0);
    dst[0] = *(const float4*)&sout[r][c0];
    dst[1] = *(const float4*)&sout[r][c0 + 4];
  } else {
    int per = ncols_store >> 2;             // 10 float4 per row for 40 cols
    if (t < 16 * per) {
      int r  = t / per;
      int c0 = (t % per) * 4;
      *(float4*)(Y + (size_t)(row0 + r) * ncols_store + c0) =
          *(const float4*)&sout[r][c0];
    }
  }
}

// ---------------------------------------------------------------------------
// Degree / normalization: deg starts at 1 (self loop), +1 per in-edge,
// then dinv = rsqrt(deg) in place.
// ---------------------------------------------------------------------------
__global__ void deg_init(float* deg, int n) {
  int i = blockIdx.x * blockDim.x + threadIdx.x;
  if (i < n) deg[i] = 1.0f;
}
__global__ void deg_count(const int* __restrict__ col, float* deg, int e) {
  int i = blockIdx.x * blockDim.x + threadIdx.x;
  if (i < e) atomicAdd(&deg[col[i]], 1.0f);
}
__global__ void deg_finish(float* deg, int n) {
  int i = blockIdx.x * blockDim.x + threadIdx.x;
  if (i < n) deg[i] = rsqrtf(deg[i]);
}

// out[n][:] = dinv[n]^2 * h[n][:]   (self-loop term; also serves as zero-init)
__global__ void prop_init(const float* __restrict__ h, const float* __restrict__ dinv,
                          float* __restrict__ out, int n, int F) {
  int i = blockIdx.x * blockDim.x + threadIdx.x;   // over n * F/4
  int q = F >> 2;
  if (i >= n * q) return;
  int node = i / q;
  float d = dinv[node];
  float w = d * d;
  float4 v = ((const float4*)h)[i];
  float4 o; o.x = w * v.x; o.y = w * v.y; o.z = w * v.z; o.w = w * v.w;
  ((float4*)out)[i] = o;
}

// one wave per edge:  out[c] += dinv[r]*dinv[c] * h[r]
// edge id forced into an SGPR so edge/dinv loads and address math scalarize.
__global__ __launch_bounds__(256) void prop_edges(
    const int* __restrict__ row, const int* __restrict__ col,
    const float* __restrict__ dinv, const float* __restrict__ h,
    float* __restrict__ out, int e, int F) {
  int g = blockIdx.x * blockDim.x + threadIdx.x;
  int eid = g >> 5, lane = g & 31;
  if (eid >= e) return;
  eid = __builtin_amdgcn_readfirstlane(eid);       // wave-uniform by construction
  int r = __builtin_amdgcn_readfirstlane(row[eid]);
  int c = __builtin_amdgcn_readfirstlane(col[eid]);
  float w = dinv[r] * dinv[c];
  const float* hs = h + (size_t)r * F;
  float*       od = out + (size_t)c * F;
  for (int f = lane * 4; f < F; f += 128) {
    float4 v = *(const float4*)(hs + f);
    atomicAdd(od + f + 0, w * v.x);
    atomicAdd(od + f + 1, w * v.y);
    atomicAdd(od + f + 2, w * v.z);
    atomicAdd(od + f + 3, w * v.w);
  }
}

// out = relu( (agg + clip(bm)*b - rm) * rsqrt(rv+eps) * g + be ), F = 128
__global__ void bias_bn_relu(const float* __restrict__ agg,
                             const float* __restrict__ b,  const float* __restrict__ bm,
                             const float* __restrict__ g,  const float* __restrict__ be,
                             const float* __restrict__ rm, const float* __restrict__ rv,
                             float* __restrict__ out, int n) {
  int i = blockIdx.x * blockDim.x + threadIdx.x;   // over n * 32
  if (i >= n * 32) return;
  int ch = (i & 31) * 4;
  float4 v = ((const float4*)agg)[i];
  float4 o;
#pragma unroll
  for (int k = 0; k < 4; ++k) {
    int c = ch + k;
    float bias = clamp01(bm[c]) * b[c];
    float s    = rsqrtf(rv[c] + BN_EPS) * g[c];
    float val  = ((&v.x)[k] + bias - rm[c]) * s + be[c];
    (&o.x)[k]  = fmaxf(val, 0.f);
  }
  ((float4*)out)[i] = o;
}

// one wave per row: add bias, then log_softmax over 40 classes
__global__ __launch_bounds__(256) void bias_logsoftmax(
    const float* __restrict__ agg, const float* __restrict__ b2,
    const float* __restrict__ bm2, float* __restrict__ out, int n) {
  int g = blockIdx.x * blockDim.x + threadIdx.x;
  int rowid = g >> 5, lane = g & 31;
  if (rowid >= n) return;
  const float* a = agg + (size_t)rowid * NC;
  int c1 = lane + 32;
  float v0 = a[lane] + clamp01(bm2[lane]) * b2[lane];
  float v1 = -3.4e38f;
  if (c1 < NC) v1 = a[c1] + clamp01(bm2[c1]) * b2[c1];
  float m = fmaxf(v0, v1);
#pragma unroll
  for (int off = 16; off; off >>= 1) m = fmaxf(m, __shfl_xor(m, off, 32));
  float s = expf(v0 - m) + ((c1 < NC) ? expf(v1 - m) : 0.f);
#pragma unroll
  for (int off = 16; off; off >>= 1) s += __shfl_xor(s, off, 32);
  float lse = m + logf(s);
  out[(size_t)rowid * NC + lane] = v0 - lse;
  if (c1 < NC) out[(size_t)rowid * NC + c1] = v1 - lse;
}

// ---------------------------------------------------------------------------
extern "C" void kernel_launch(void* const* d_in, const int* in_sizes, int n_in,
                              void* d_out, int out_size, void* d_ws, size_t ws_size,
                              hipStream_t stream) {
  const float* x   = (const float*)d_in[0];
  const int*   ei  = (const int*)  d_in[1];
  const float* W0  = (const float*)d_in[2],  *Wm0 = (const float*)d_in[3];
  const float* b0  = (const float*)d_in[4],  *bm0 = (const float*)d_in[5];
  const float* g0  = (const float*)d_in[6],  *be0 = (const float*)d_in[7];
  const float* rm0 = (const float*)d_in[8],  *rv0 = (const float*)d_in[9];
  const float* W1  = (const float*)d_in[10], *Wm1 = (const float*)d_in[11];
  const float* b1  = (const float*)d_in[12], *bm1 = (const float*)d_in[13];
  const float* g1  = (const float*)d_in[14], *be1 = (const float*)d_in[15];
  const float* rm1 = (const float*)d_in[16], *rv1 = (const float*)d_in[17];
  const float* W2  = (const float*)d_in[18], *Wm2 = (const float*)d_in[19];
  const float* b2  = (const float*)d_in[20], *bm2 = (const float*)d_in[21];

  const int n = in_sizes[0] / FIN;          // 100000
  const int e = in_sizes[1] / 2;            // 1600000
  const int* erow = ei;                     // source (x_j)
  const int* ecol = ei + e;                 // aggregation target

  char* ws = (char*)d_ws;
  auto take = [&](size_t bytes) -> char* {
    char* p = ws; ws += (bytes + 255) & ~(size_t)255; return p;
  };
  float*     dinv = (float*)take((size_t)n * 4);
  float*     bufA = (float*)take((size_t)n * HD * 4);
  float*     bufB = (float*)take((size_t)n * HD * 4);
  float*     bufL = (float*)take((size_t)n * NC * 4);
  float*     bufM = (float*)take((size_t)n * NC * 4);
  _Float16*  Bp0  = (_Float16*)take(4 * 8 * 512 * 2);
  _Float16*  Bp1  = (_Float16*)take(4 * 8 * 512 * 2);
  _Float16*  Bp2  = (_Float16*)take(4 * 3 * 512 * 2);

  const int TB = 256;
  // masked weights -> f16 B-fragments
  pack_w<<<(4 * 8 * 512 + TB - 1) / TB, TB, 0, stream>>>(W0, Wm0, Bp0, 128, 8);
  pack_w<<<(4 * 8 * 512 + TB - 1) / TB, TB, 0, stream>>>(W1, Wm1, Bp1, 128, 8);
  pack_w<<<(4 * 3 * 512 + TB - 1) / TB, TB, 0, stream>>>(W2, Wm2, Bp2, NC, 3);

  // symmetric normalization coefficients
  deg_init  <<<(n + TB - 1) / TB, TB, 0, stream>>>(dinv, n);
  deg_count <<<(e + TB - 1) / TB, TB, 0, stream>>>(ecol, dinv, e);
  deg_finish<<<(n + TB - 1) / TB, TB, 0, stream>>>(dinv, n);

  // layer 0
  gemm_wmma <<<n / 16, TB, 0, stream>>>(x, Bp0, bufA, 128, 8);
  prop_init <<<(n * 32 + TB - 1) / TB, TB, 0, stream>>>(bufA, dinv, bufB, n, 128);
  prop_edges<<<(e * 32 + TB - 1) / TB, TB, 0, stream>>>(erow, ecol, dinv, bufA, bufB, e, 128);
  bias_bn_relu<<<(n * 32 + TB - 1) / TB, TB, 0, stream>>>(bufB, b0, bm0, g0, be0, rm0, rv0, bufA, n);

  // layer 1
  gemm_wmma <<<n / 16, TB, 0, stream>>>(bufA, Bp1, bufB, 128, 8);
  prop_init <<<(n * 32 + TB - 1) / TB, TB, 0, stream>>>(bufB, dinv, bufA, n, 128);
  prop_edges<<<(e * 32 + TB - 1) / TB, TB, 0, stream>>>(erow, ecol, dinv, bufB, bufA, e, 128);
  bias_bn_relu<<<(n * 32 + TB - 1) / TB, TB, 0, stream>>>(bufA, b1, bm1, g1, be1, rm1, rv1, bufB, n);

  // layer 2 + log_softmax
  gemm_wmma <<<n / 16, TB, 0, stream>>>(bufB, Bp2, bufL, NC, 3);
  prop_init <<<(n * (NC / 4) + TB - 1) / TB, TB, 0, stream>>>(bufL, dinv, bufM, n, NC);
  prop_edges<<<(e * 32 + TB - 1) / TB, TB, 0, stream>>>(erow, ecol, dinv, bufL, bufM, e, NC);
  bias_logsoftmax<<<(n * 32 + TB - 1) / TB, TB, 0, stream>>>(bufM, b2, bm2, (float*)d_out, n);
}
